// LatentAttentionBlock_24154896073285
// MI455X (gfx1250) — matrix operand order repaired and checked by reading, not compiled, not run on hardware
//
#include <hip/hip_runtime.h>
#include <hip/hip_bf16.h>
#include <math.h>

#define WIDTH   1024
#define NHEADS  8
#define QKD     128     // total qk dim
#define HQK     16      // per-head qk dim
#define EXPD    2048
#define VD      1024
#define HV      128     // per-head v dim
#define HID     4352    // 2*QKD + 2*EXPD
#define BB      2
#define SS      2048
#define NT      4096    // BB*SS tokens

typedef _Float16 h16;
typedef __attribute__((ext_vector_type(16))) _Float16 v16h;
typedef __attribute__((ext_vector_type(8)))  float    v8f;

union Frag16 { v16h h; uint4 u[2]; h16 e[16]; };

// XOR-butterfly via ds_swizzle bitmask mode: src_lane = ((lane & 0x1f) | 0) ^ m
#define SWZ(v, m) __int_as_float(__builtin_amdgcn_ds_swizzle(__float_as_int(v), (((m) << 10) | 0x1f)))

// ---- async global->LDS staging (CDNA5 GLOBAL_LOAD_ASYNC_TO_LDS_B128) ------
#if __has_builtin(__builtin_amdgcn_global_load_async_to_lds_b128) && \
    __has_builtin(__builtin_amdgcn_s_wait_asynccnt)
#define USE_ASYNC 1
typedef int v4i __attribute__((vector_size(16)));       // matches builtin proto
typedef __attribute__((address_space(1))) v4i* as1_v4ip;
typedef __attribute__((address_space(3))) v4i* as3_v4ip;
__device__ __forceinline__ void async_copy16(h16* lds, const h16* g) {
  // generic LDS VA carries the LDS byte offset in its low 32 bits; global flat
  // VA is identity-mapped -> reinterpret through uintptr_t is value-preserving
  // (and drops const, as the builtin takes non-const pointers).
  __builtin_amdgcn_global_load_async_to_lds_b128(
      (as1_v4ip)(uintptr_t)g, (as3_v4ip)(uintptr_t)lds, 0, 0);
}
#else
#define USE_ASYNC 0
#endif

// ---- WMMA fragment loaders (layouts per CDNA5 ISA 7.12.2) -----------------
// A (16x32 f16): lane row = lane&15; lanes<16 hold K 0..7 & 16..23,
// lanes>=16 hold K 8..15 & 24..31 (pairs packed per VGPR).
__device__ __forceinline__ v16h load_a_frag(const h16* t, int stride) {
  int lane = threadIdx.x & 31;
  int row  = lane & 15;
  int kb   = (lane < 16) ? 0 : 8;
  Frag16 f;
  f.u[0] = *(const uint4*)(t + row * stride + kb);
  f.u[1] = *(const uint4*)(t + row * stride + kb + 16);
  return f.h;
}
// B (32x16 f16) from LDS tile stored as [N=16 rows][K=32 cols]:
// lane col = lane&15; lanes<16 hold K 0..7 & 16..23, lanes>=16 hold K 8..15 & 24..31.
__device__ __forceinline__ v16h load_b_frag(const h16* t, int stride) {
  int lane = threadIdx.x & 31;
  int col  = lane & 15;
  int kb   = (lane < 16) ? 0 : 8;
  Frag16 f;
  f.u[0] = *(const uint4*)(t + col * stride + kb);
  f.u[1] = *(const uint4*)(t + col * stride + kb + 16);
  return f.h;
}

// ---- prep kernels ----------------------------------------------------------
__global__ void __launch_bounds__(256)
cast_f32_f16(const float* __restrict__ in, h16* __restrict__ out, int n) {
  int i = blockIdx.x * 256 + threadIdx.x;
  if (i < n) out[i] = (h16)in[i];
}

// expand weights: fold attention scale 1/sqrt(HQK)=0.25 into the q rows (0..127)
__global__ void __launch_bounds__(256)
cast_expand_w(const float* __restrict__ in, h16* __restrict__ out, int n) {
  int i = blockIdx.x * 256 + threadIdx.x;
  if (i < n) {
    float v = in[i];
    if (i < QKD * WIDTH) v *= 0.25f;
    out[i] = (h16)v;
  }
}

__global__ void __launch_bounds__(256)
ln_kernel(const float* __restrict__ x, const float* __restrict__ w,
          h16* __restrict__ xh) {
  int row = blockIdx.x;
  int t = threadIdx.x;
  const float* xr = x + (size_t)row * WIDTH;
  float4 xv = *(const float4*)(xr + t * 4);
  float s  = xv.x + xv.y + xv.z + xv.w;
  float s2 = xv.x*xv.x + xv.y*xv.y + xv.z*xv.z + xv.w*xv.w;
#pragma unroll
  for (int off = 16; off > 0; off >>= 1) {
    s  += __shfl_xor(s,  off, 32);
    s2 += __shfl_xor(s2, off, 32);
  }
  __shared__ float red[16];
  int lane = t & 31, wid = t >> 5;
  if (lane == 0) { red[wid] = s; red[8 + wid] = s2; }
  __syncthreads();
  float S = 0.f, S2 = 0.f;
#pragma unroll
  for (int i = 0; i < 8; ++i) { S += red[i]; S2 += red[8 + i]; }
  float mu  = S * (1.0f / WIDTH);
  float var = S2 * (1.0f / WIDTH) - mu * mu;
  float r   = rsqrtf(var + 1e-5f);
  float4 wv = *(const float4*)(w + t * 4);
  h16* op = xh + (size_t)row * WIDTH + t * 4;
  op[0] = (h16)((xv.x - mu) * r * wv.x);
  op[1] = (h16)((xv.y - mu) * r * wv.y);
  op[2] = (h16)((xv.z - mu) * r * wv.z);
  op[3] = (h16)((xv.w - mu) * r * wv.w);
}

// ---- generic WMMA GEMM: C[M,N] = A[M,K] * W[N,K]^T -------------------------
// block tile 256x128, 8 waves x (32 M rows, 128 N cols), K step 32,
// double-buffered LDS staging (one barrier per K step), 16 WMMA / 10 frags.
// Staging uses async global->LDS copies when available (no VGPR round trip).
template <bool RESID>
__global__ void __launch_bounds__(256)
gemm_wmma(const h16* __restrict__ A, int lda,
          const h16* __restrict__ Bw, int K,
          h16* __restrict__ Ch, float* __restrict__ Cf,
          const float* __restrict__ xres, int ldc) {
  __shared__ __align__(16) h16 As[2][256 * 40];
  __shared__ __align__(16) h16 Bs[2][128 * 40];
  int m0 = blockIdx.y * 256;
  int n0 = blockIdx.x * 128;
  int t = threadIdx.x, lane = t & 31, wid = t >> 5;

  v8f acc[16];
#pragma unroll
  for (int g = 0; g < 16; ++g)
#pragma unroll
    for (int i = 0; i < 8; ++i) acc[g][i] = 0.0f;

#if USE_ASYNC
  auto astage = [&](int k0, int buf) {
#pragma unroll
    for (int q4 = 0; q4 < 4; ++q4) {
      int c = t + q4 * 256, r = c >> 2, sg = (c & 3) * 8;
      async_copy16(As[buf] + r * 40 + sg, A + (size_t)(m0 + r) * lda + k0 + sg);
    }
#pragma unroll
    for (int q2 = 0; q2 < 2; ++q2) {
      int c = t + q2 * 256, r = c >> 2, sg = (c & 3) * 8;
      async_copy16(Bs[buf] + r * 40 + sg, Bw + (size_t)(n0 + r) * K + k0 + sg);
    }
  };
  astage(0, 0);
#else
  uint4 ra[4], rb[2];
  auto gload = [&](int k0) {
#pragma unroll
    for (int q4 = 0; q4 < 4; ++q4) {
      int c = t + q4 * 256, r = c >> 2, sg = (c & 3) * 8;
      ra[q4] = *(const uint4*)(A + (size_t)(m0 + r) * lda + k0 + sg);
    }
#pragma unroll
    for (int q2 = 0; q2 < 2; ++q2) {
      int c = t + q2 * 256, r = c >> 2, sg = (c & 3) * 8;
      rb[q2] = *(const uint4*)(Bw + (size_t)(n0 + r) * K + k0 + sg);
    }
  };
  auto sstore = [&](int buf) {
#pragma unroll
    for (int q4 = 0; q4 < 4; ++q4) {
      int c = t + q4 * 256, r = c >> 2, sg = (c & 3) * 8;
      *(uint4*)(As[buf] + r * 40 + sg) = ra[q4];
    }
#pragma unroll
    for (int q2 = 0; q2 < 2; ++q2) {
      int c = t + q2 * 256, r = c >> 2, sg = (c & 3) * 8;
      *(uint4*)(Bs[buf] + r * 40 + sg) = rb[q2];
    }
  };
  gload(0);
  sstore(0);
#endif

#if USE_ASYNC
  __builtin_amdgcn_s_wait_asynccnt(0);
#endif
  __syncthreads();

  int nkt = K >> 5;
  for (int kt = 0; kt < nkt; ++kt) {
    int cur = kt & 1;
    if (kt + 2 < nkt) {                         // pull K+2 toward L2
      __builtin_prefetch(A  + (size_t)(m0 + t) * lda + (kt + 2) * 32, 0, 1);
      __builtin_prefetch(Bw + (size_t)(n0 + (t & 127)) * K + (kt + 2) * 32, 0, 1);
    }
#if USE_ASYNC
    if (kt + 1 < nkt) astage((kt + 1) * 32, 1 - cur);  // async into other buffer
#else
    if (kt + 1 < nkt) gload((kt + 1) * 32);     // overlap with WMMAs below
#endif
    v16h a0 = load_a_frag(As[cur] + (wid * 32) * 40, 40);
    v16h a1 = load_a_frag(As[cur] + (wid * 32 + 16) * 40, 40);
#pragma unroll
    for (int g = 0; g < 8; ++g) {
      v16h b = load_b_frag(Bs[cur] + g * 16 * 40, 40);
      acc[g]     = __builtin_amdgcn_wmma_f32_16x16x32_f16(false, a0, false, b,
                                                          (short)0, acc[g], false, false);
      acc[8 + g] = __builtin_amdgcn_wmma_f32_16x16x32_f16(false, a1, false, b,
                                                          (short)0, acc[8 + g], false, false);
    }
#if USE_ASYNC
    __builtin_amdgcn_s_wait_asynccnt(0);        // this wave's copies landed
#else
    if (kt + 1 < nkt) sstore(1 - cur);          // write the other buffer
#endif
    __syncthreads();                            // publish tile block-wide
  }

  int col16 = lane & 15, rowb = (lane < 16) ? 0 : 8;
#pragma unroll
  for (int half = 0; half < 2; ++half)
#pragma unroll
    for (int g = 0; g < 8; ++g)
#pragma unroll
      for (int i = 0; i < 8; ++i) {
        size_t row = (size_t)(m0 + wid * 32 + half * 16 + rowb + i);
        int    col = n0 + g * 16 + col16;
        float  v   = acc[half * 8 + g][i];
        if (RESID) Cf[row * ldc + col] = xres[row * ldc + col] + v;
        else       Ch[row * ldc + col] = (h16)v;
      }
}

// ---- GEGLU split -----------------------------------------------------------
__global__ void __launch_bounds__(256)
geglu_kernel(const h16* __restrict__ H, h16* __restrict__ U,
             h16* __restrict__ Vb) {
  size_t i = (size_t)blockIdx.x * 256 + threadIdx.x;   // over NT*EXPD
  if (i >= (size_t)NT * EXPD) return;
  size_t tok = i >> 11;
  int j = (int)(i & 2047);
  float lin = (float)H[tok * HID + 256 + j];
  float pre = (float)H[tok * HID + 2304 + j];
  float ge  = 0.5f * pre * (1.0f + erff(pre * 0.70710678118f));  // exact GELU
  float g   = lin * ge;
  if (j < 1024) U[tok * 2048 + j]           = (h16)g;
  else          Vb[tok * 1024 + (j - 1024)] = (h16)g;
}

// ---- flash-style causal attention ------------------------------------------
// grid: (S/128 qtiles, NHEADS, B); block: 256 = 8 waves; wave -> 16 q rows.
// V tiles double-buffered in LDS; scale pre-folded into q weights;
// row reductions via single-instruction ds_swizzle XOR butterflies.
__global__ void __launch_bounds__(256)
attn_kernel(const h16* __restrict__ H, const h16* __restrict__ Vb,
            h16* __restrict__ U, const float* __restrict__ pos_mult) {
  __shared__ __align__(16) h16 VT[2][128 * 40];   // V^T tile: [d 0..127][key 0..31]
  __shared__ __align__(16) h16 PB[8][16 * 40];    // per-wave P staging (C -> A layout)
  int q0 = blockIdx.x * 128;
  int h  = blockIdx.y;
  int b  = blockIdx.z;
  int t = threadIdx.x, lane = t & 31, wid = t >> 5;
  int qw = q0 + wid * 16;
  int col16 = lane & 15, rowb = (lane < 16) ? 0 : 8;
  float sp = log1pf(__expf(pos_mult[0]));         // softplus(pos_mult)
  float spi[8];                                   // sp * i table
#pragma unroll
  for (int i = 0; i < 8; ++i) spi[i] = sp * (float)i;

  // Q A-fragment: per-head K dim = 16, zero-padded to 32. (scale pre-folded)
  Frag16 qa;
  qa.u[1] = make_uint4(0, 0, 0, 0);
  {
    int row = qw + (lane & 15);
    int kb  = (lane < 16) ? 0 : 8;
    qa.u[0] = *(const uint4*)(H + ((size_t)(b * SS) + row) * HID + h * HQK + kb);
  }

  v8f acc[8];
  float m_[8], l_[8];
#pragma unroll
  for (int g = 0; g < 8; ++g)
#pragma unroll
    for (int i = 0; i < 8; ++i) acc[g][i] = 0.0f;
#pragma unroll
  for (int i = 0; i < 8; ++i) { m_[i] = -1e30f; l_[i] = 0.0f; }

  v8f vzero;
#pragma unroll
  for (int i = 0; i < 8; ++i) vzero[i] = 0.0f;

  int vj = t >> 3, vdb = (t & 7) * 16;            // V staging assignment
  Frag16 vf;
  auto vload = [&](int jt) {
    const h16* src = Vb + ((size_t)(b * SS) + jt + vj) * VD + h * HV + vdb;
    vf.u[0] = *(const uint4*)(src);
    vf.u[1] = *(const uint4*)(src + 8);
  };
  auto vstore = [&](int buf) {
#pragma unroll
    for (int i = 0; i < 16; ++i) VT[buf][(vdb + i) * 40 + vj] = vf.e[i];
  };

  int njt = ((q0 + 127) >> 5) + 1;                // # of 32-key tiles
  vload(0);
  vstore(0);
  __syncthreads();

  for (int it = 0; it < njt; ++it) {
    int jt = it * 32;
    int cur = it & 1;
    if (it + 1 < njt) vload((it + 1) * 32);       // overlap with compute

    if (jt <= qw + 15) {                          // wave-uniform skip
      // K B-fragments for keys jt..jt+15 / jt+16..jt+31 (feature K zero-padded)
      Frag16 kf0, kf1;
      kf0.u[0] = kf0.u[1] = kf1.u[0] = kf1.u[1] = make_uint4(0, 0, 0, 0);
      {
        const h16* kp = H + ((size_t)(b * SS) + jt + col16) * HID + QKD + h * HQK + rowb;
        kf0.u[0] = *(const uint4*)(kp);
        const h16* kp1 = kp + (size_t)16 * HID;
        kf1.u[0] = *(const uint4*)(kp1);
      }
      v8f s0 = __builtin_amdgcn_wmma_f32_16x16x32_f16(false, qa.h, false, kf0.h,
                                                      (short)0, vzero, false, false);
      v8f s1 = __builtin_amdgcn_wmma_f32_16x16x32_f16(false, qa.h, false, kf1.h,
                                                      (short)0, vzero, false, false);
      h16* pb = PB[wid];
      // bias is affine in row index i: sp*(k - q) = f0 - sp*i; mask: d0 <= i
      int   d0A = jt + col16 - qw - rowb;
      int   d0B = d0A + 16;
      float f0A = sp * (float)d0A;
      float f0B = f0A + sp * 16.0f;
#pragma unroll
      for (int i = 0; i < 8; ++i) {
        float b0 = (d0A <= i) ? s0[i] + (f0A - spi[i]) : -1e30f;
        float b1 = (d0B <= i) ? s1[i] + (f0B - spi[i]) : -1e30f;
        float tm = fmaxf(b0, b1);
        tm = fmaxf(tm, SWZ(tm, 1));               // stays within 16-lane row group
        tm = fmaxf(tm, SWZ(tm, 2));
        tm = fmaxf(tm, SWZ(tm, 4));
        tm = fmaxf(tm, SWZ(tm, 8));
        float mnew = fmaxf(m_[i], tm);
        float a  = __expf(m_[i] - mnew);
        float e0 = __expf(b0 - mnew);
        float e1 = __expf(b1 - mnew);
        float ts = e0 + e1;
        ts += SWZ(ts, 1);
        ts += SWZ(ts, 2);
        ts += SWZ(ts, 4);
        ts += SWZ(ts, 8);
        l_[i] = l_[i] * a + ts;
        m_[i] = mnew;
        pb[(rowb + i) * 40 + col16]      = (h16)e0;
        pb[(rowb + i) * 40 + 16 + col16] = (h16)e1;
#pragma unroll
        for (int g = 0; g < 8; ++g) acc[g][i] *= a;
      }
      v16h pa = load_a_frag(pb, 40);
#pragma unroll
      for (int g = 0; g < 8; ++g) {
        v16h vb2 = load_b_frag(VT[cur] + g * 16 * 40, 40);
        acc[g] = __builtin_amdgcn_wmma_f32_16x16x32_f16(false, pa, false, vb2,
                                                        (short)0, acc[g], false, false);
      }
    }

    if (it + 1 < njt) vstore(1 - cur);
    __syncthreads();
  }

#pragma unroll
  for (int g = 0; g < 8; ++g)
#pragma unroll
    for (int i = 0; i < 8; ++i) {
      int q = qw + rowb + i;
      float o = acc[g][i] / l_[i];
      U[((size_t)(b * SS) + q) * 2048 + 1024 + h * HV + g * 16 + col16] = (h16)o;
    }
}

// ---- launcher --------------------------------------------------------------
extern "C" void kernel_launch(void* const* d_in, const int* in_sizes, int n_in,
                              void* d_out, int out_size, void* d_ws, size_t ws_size,
                              hipStream_t stream) {
  const float* x         = (const float*)d_in[0];
  // d_in[1] attn_mask, d_in[2] pos_bias: computed analytically, not read
  const float* norm_w    = (const float*)d_in[3];
  const float* expand_w  = (const float*)d_in[4];
  const float* project_w = (const float*)d_in[5];
  const float* pos_mult  = (const float*)d_in[6];
  float* out = (float*)d_out;

  char* ws = (char*)d_ws;
  h16* Xh = (h16*)(ws);                       //  8,388,608 B  [4096,1024] f16
  h16* We = (h16*)(ws + 8388608);             //  8,912,896 B  [4352,1024] f16
  h16* Wp = (h16*)(ws + 17301504);            //  4,194,304 B  [1024,2048] f16
  h16* Hb = (h16*)(ws + 21495808);            // 35,651,584 B  [4096,4352] f16
  h16* U  = (h16*)(ws + 57147392);            // 16,777,216 B  [4096,2048] f16
  h16* Vb = (h16*)(ws + 73924608);            //  8,388,608 B  [4096,1024] f16

  cast_expand_w<<<(4456448 + 255) / 256, 256, 0, stream>>>(expand_w, We, 4456448);
  cast_f32_f16<<<(2097152 + 255) / 256, 256, 0, stream>>>(project_w, Wp, 2097152);
  ln_kernel<<<NT, 256, 0, stream>>>(x, norm_w, Xh);
  // hidden = Xn @ expand_w^T  : [4096,1024] x [4352,1024]^T
  gemm_wmma<false><<<dim3(34, 16), 256, 0, stream>>>(Xh, 1024, We, 1024,
                                                     Hb, nullptr, nullptr, HID);
  geglu_kernel<<<(NT * EXPD + 255) / 256, 256, 0, stream>>>(Hb, U, Vb);
  attn_kernel<<<dim3(16, NHEADS, BB), 256, 0, stream>>>(Hb, Vb, U, pos_mult);
  // out = x + [geglu_local | attn_out] @ project_w^T : [4096,2048] x [1024,2048]^T
  gemm_wmma<true><<<dim3(8, 16), 256, 0, stream>>>(U, 2048, Wp, 2048,
                                                   nullptr, out, x, 1024);
}
